// TimeSeriesSDE_52072183497422
// MI455X (gfx1250) — matrix operand order repaired and checked by reading, not compiled, not run on hardware
//
#include <hip/hip_runtime.h>
#include <hip/hip_bf16.h>
#include <math.h>

typedef __bf16 bf16_t;
typedef __bf16 bf16x8  __attribute__((ext_vector_type(8)));
typedef __bf16 bf16x16 __attribute__((ext_vector_type(16)));
typedef float  f32x8   __attribute__((ext_vector_type(8)));

#define TOTAL_STEPS 108
#define NSUB 12
#define EPT 8                 // state elements per thread (16*128 / 256)
#define ROWS_PER_WG 16

// Branch-free native-HW activations (v_exp_f32 / v_log_f32 / v_rcp_f32).
// tanh(x) = 1 - 2/(exp(2x)+1)  (exact identity; overflow-safe: e->inf => 1)
__device__ __forceinline__ float fast_tanh(float x) {
  float e = __expf(2.0f * x);
  float r = __builtin_amdgcn_rcpf(e + 1.0f);   // single v_rcp_f32
  return 1.0f - 2.0f * r;
}
// softplus(x) = log(1+exp(x)); for large x select x (e=inf would give inf)
__device__ __forceinline__ float fast_softplus(float x) {
  float l = __logf(1.0f + __expf(x));
  return (x > 20.f) ? x : l;
}

// ---------------------------------------------------------------------------
// GEMM with per-wave register-resident B tiles.
// A: LDS row-major bf16 [16][KD]. Breg: [KT*NPW] WMMA-B operands.
// ACT: 0=tanh->bf16, 1=linear->f32, 2=softplus->f32. 8 waves, NPW=ND/128.
// ---------------------------------------------------------------------------
template <int KD, int ND, int ACT>
__device__ __forceinline__ void gemm_reg(const bf16_t* __restrict__ A,
                                         const bf16x16* __restrict__ Breg,
                                         const float* __restrict__ biasv,
                                         bf16_t* __restrict__ outB,
                                         float* __restrict__ outF) {
  const int tid = threadIdx.x;
  const int lane = tid & 31;
  const int wave = tid >> 5;
  const int half = lane >> 4;
  const int mrow = lane & 15;
  const int nl = lane & 15;
  constexpr int KT = KD / 32, NT = ND / 16, NPW = NT / 8;
  const int nt0 = wave * NPW;
  f32x8 c[NPW] = {};
#pragma unroll
  for (int kt = 0; kt < KT; ++kt) {
    const bf16_t* ap = A + mrow * KD + kt * 32 + half * 8;
    bf16x8 lo = *(const bf16x8*)ap;
    bf16x8 hi = *(const bf16x8*)(ap + 16);
    bf16x16 a = __builtin_shufflevector(lo, hi, 0, 1, 2, 3, 4, 5, 6, 7,
                                        8, 9, 10, 11, 12, 13, 14, 15);
#pragma unroll
    for (int p = 0; p < NPW; ++p)
      c[p] = __builtin_amdgcn_wmma_f32_16x16x32_bf16(false, a, false,
                                                     Breg[kt * NPW + p],
                                                     (short)0, c[p], false, false);
  }
#pragma unroll
  for (int p = 0; p < NPW; ++p) {
    const int Ncol = (nt0 + p) * 16 + nl;
    const float bs = biasv[p];
#pragma unroll
    for (int r = 0; r < 8; ++r) {
      const int m = r + half * 8;
      float v = c[p][r] + bs;
      if constexpr (ACT == 0)      outB[m * ND + Ncol] = (bf16_t)fast_tanh(v);
      else if constexpr (ACT == 1) outF[m * ND + Ncol] = v;
      else                         outF[m * ND + Ncol] = fast_softplus(v);
    }
  }
}

// Same but B streamed from global packed tiles (used for the 256x256 W2).
template <int KD, int ND, int ACT>
__device__ __forceinline__ void gemm_glb(const bf16_t* __restrict__ A,
                                         const unsigned* __restrict__ Bp,
                                         const float* __restrict__ biasv,
                                         bf16_t* __restrict__ outB,
                                         float* __restrict__ outF) {
  const int tid = threadIdx.x;
  const int lane = tid & 31;
  const int wave = tid >> 5;
  const int half = lane >> 4;
  const int mrow = lane & 15;
  const int nl = lane & 15;
  constexpr int KT = KD / 32, NT = ND / 16, NPW = NT / 8;
  const int nt0 = wave * NPW;
  f32x8 c[NPW] = {};
#pragma unroll
  for (int kt = 0; kt < KT; ++kt) {
    const bf16_t* ap = A + mrow * KD + kt * 32 + half * 8;
    bf16x8 lo = *(const bf16x8*)ap;
    bf16x8 hi = *(const bf16x8*)(ap + 16);
    bf16x16 a = __builtin_shufflevector(lo, hi, 0, 1, 2, 3, 4, 5, 6, 7,
                                        8, 9, 10, 11, 12, 13, 14, 15);
#pragma unroll
    for (int p = 0; p < NPW; ++p) {
      bf16x16 b = *(const bf16x16*)(Bp + (((kt * NT) + nt0 + p) << 8) + (lane << 3));
      c[p] = __builtin_amdgcn_wmma_f32_16x16x32_bf16(false, a, false, b,
                                                     (short)0, c[p], false, false);
    }
  }
#pragma unroll
  for (int p = 0; p < NPW; ++p) {
    const int Ncol = (nt0 + p) * 16 + nl;
    const float bs = biasv[p];
#pragma unroll
    for (int r = 0; r < 8; ++r) {
      const int m = r + half * 8;
      float v = c[p][r] + bs;
      if constexpr (ACT == 0)      outB[m * ND + Ncol] = (bf16_t)fast_tanh(v);
      else if constexpr (ACT == 1) outF[m * ND + Ncol] = v;
      else                         outF[m * ND + Ncol] = fast_softplus(v);
    }
  }
}

// ---------------------------------------------------------------------------
// One-time weight pack: fp32 row-major [K][N] -> bf16 WMMA-B tiles.
// tile(kt,nt)=256 dw; lane L dword j = (k=kt*32+(L>=16?8:0)+2j+(j>=4?8:0), n=nt*16+(L&15))
// ---------------------------------------------------------------------------
__global__ void pack_weight(const float* __restrict__ W, unsigned* __restrict__ P,
                            int K, int N) {
  int d = blockIdx.x * blockDim.x + threadIdx.x;
  int total = (K * N) >> 1;
  if (d >= total) return;
  int NT = N >> 4;
  int tile = d >> 8;
  int within = d & 255;
  int lane = within >> 3;
  int j = within & 7;
  int kt = tile / NT;
  int nt = tile - kt * NT;
  int n = (nt << 4) + (lane & 15);
  int k = (kt << 5) + ((lane >> 4) << 3) + (j << 1) + ((j >> 2) << 3);
  union { bf16_t h[2]; unsigned u; } pk;
  pk.h[0] = (bf16_t)W[k * N + n];
  pk.h[1] = (bf16_t)W[(k + 1) * N + n];
  P[d] = pk.u;
}

// ---------------------------------------------------------------------------
// Persistent SRK-SDE kernel: 64 WGs x 256 threads, 16 rows/WG, 108 steps.
// W1/V1/W3/V2 tiles live in per-wave VGPRs; W2 streams from L2/WGP$.
// ---------------------------------------------------------------------------
__global__ __launch_bounds__(256) void sde_kernel(
    const float* __restrict__ x,
    const float* __restrict__ b1, const float* __restrict__ b2,
    const float* __restrict__ b3, const float* __restrict__ c1,
    const float* __restrict__ c2,
    const float* __restrict__ zw, const float* __restrict__ zl,
    const unsigned* __restrict__ W1p, const unsigned* __restrict__ W2p,
    const unsigned* __restrict__ W3p, const unsigned* __restrict__ V1p,
    const unsigned* __restrict__ V2p, float* __restrict__ out) {
  __shared__ __attribute__((aligned(16))) bf16_t sA[16 * 128];   // drift A / shared A
  __shared__ __attribute__((aligned(16))) bf16_t sAd[16 * 128];  // diff A (stage 1)
  __shared__ __attribute__((aligned(16))) bf16_t sH[16 * 256];   // drift hidden1
  __shared__ __attribute__((aligned(16))) bf16_t sHd[16 * 256];  // diff hidden
  __shared__ __attribute__((aligned(16))) bf16_t sH2[16 * 256];  // drift hidden2
  __shared__ __attribute__((aligned(16))) float  sF[16 * 128];   // drift out f32
  __shared__ __attribute__((aligned(16))) float  sG[16 * 128];   // diff  out f32

  const int tid = threadIdx.x;
  const int lane = tid & 31;
  const int wave = tid >> 5;
  const int nl = lane & 15;
  const int wg = blockIdx.x;
  const int base = tid * EPT;
  const int rowBase = wg * ROWS_PER_WG;

  const float dt = (float)(10.0 / 108.0);
  const float sqdt = sqrtf(dt);
  const float inv_dt = 1.f / dt;
  const float inv_sqdt = 1.f / sqdt;
  const float czl = dt * sqdt * 0.28867513459481287f;  // dt*sqrt(dt)/sqrt(12)

  // ---- preload per-wave register-resident weight tiles ----
  bf16x16 BW1[8], BV1[8], BW3[8], BV2[8];
#pragma unroll
  for (int kt = 0; kt < 4; ++kt)
#pragma unroll
    for (int p = 0; p < 2; ++p) {
      BW1[kt * 2 + p] = *(const bf16x16*)(W1p + ((kt * 16 + wave * 2 + p) << 8) + (lane << 3));
      BV1[kt * 2 + p] = *(const bf16x16*)(V1p + ((kt * 16 + wave * 2 + p) << 8) + (lane << 3));
    }
#pragma unroll
  for (int kt = 0; kt < 8; ++kt) {
    BW3[kt] = *(const bf16x16*)(W3p + ((kt * 8 + wave) << 8) + (lane << 3));
    BV2[kt] = *(const bf16x16*)(V2p + ((kt * 8 + wave) << 8) + (lane << 3));
  }
  // per-lane biases
  float bs1[2], bs2[2], cs1[2], bs3[1], cs2[1];
#pragma unroll
  for (int p = 0; p < 2; ++p) {
    bs1[p] = b1[(wave * 2 + p) * 16 + nl];
    bs2[p] = b2[(wave * 2 + p) * 16 + nl];
    cs1[p] = c1[(wave * 2 + p) * 16 + nl];
  }
  bs3[0] = b3[wave * 16 + nl];
  cs2[0] = c2[wave * 16 + nl];

  float y0r[EPT], y1r[EPT], f0r[EPT], g0r[EPT], g1r[EPT], g2r[EPT];
  float IkR[EPT], Ik0R[EPT], IkkR[EPT], IkkkR[EPT], tmpA[EPT], tmpB[EPT];

  // init: y0 = x[:, :, -1] broadcast over samples; checkpoint 0
#pragma unroll
  for (int e = 0; e < EPT; ++e) {
    int idx = base + e;
    int rl = idx >> 7, n = idx & 127;
    int r = rowBase + rl;
    int b = r & 255;  // r = s*256 + b
    float v = x[(b * 128 + n) * 48 + 47];
    y0r[e] = v;
    out[(size_t)(r * 128 + n) * 10 + 0] = v;
  }

  auto stage = [&](bf16_t* dst, const float* v) {
#pragma unroll
    for (int e = 0; e < EPT; ++e) dst[base + e] = (bf16_t)v[e];
  };

  for (int t = 0; t < TOTAL_STEPS; ++t) {
    const size_t nbase = (size_t)t * 131072 + (size_t)rowBase * 128 + base;
    if (t + 1 < TOTAL_STEPS) {
      __builtin_prefetch(zw + nbase + 131072, 0, 0);
      __builtin_prefetch(zl + nbase + 131072, 0, 0);
    }
#pragma unroll
    for (int e = 0; e < EPT; ++e) {
      float ik = sqdt * zw[nbase + e];
      IkR[e] = ik;
      Ik0R[e] = 0.5f * dt * ik + czl * zl[nbase + e];
      IkkR[e] = 0.5f * (ik * ik - dt);
      IkkkR[e] = (ik * ik * ik - 3.f * dt * ik) * (1.f / 6.f);
    }

    // ======== stage 0: drift+diff at y0 (shared input) ========
    stage(sA, y0r);
    __syncthreads();
    gemm_reg<128, 256, 0>(sA, BW1, bs1, sH, nullptr);
    gemm_reg<128, 256, 0>(sA, BV1, cs1, sHd, nullptr);
    __syncthreads();
    gemm_glb<256, 256, 0>(sH, W2p, bs2, sH2, nullptr);
    gemm_reg<256, 128, 2>(sHd, BV2, cs2, nullptr, sG);
    __syncthreads();
    gemm_reg<256, 128, 1>(sH2, BW3, bs3, nullptr, sF);
    __syncthreads();
#pragma unroll
    for (int e = 0; e < EPT; ++e) {
      f0r[e] = sF[base + e];
      g0r[e] = sG[base + e];
      float gw = -IkR[e] + IkkR[e] * inv_sqdt +
                 2.f * Ik0R[e] * inv_dt - 2.f * IkkkR[e] * inv_dt;
      y1r[e] = y0r[e] + (dt * (1.f / 3.f)) * f0r[e] + gw * g0r[e];
      tmpA[e] = y0r[e] + 0.75f * dt * f0r[e] + (1.5f * inv_dt) * Ik0R[e] * g0r[e];
      tmpB[e] = y0r[e] + 0.25f * dt * f0r[e] + 0.5f * sqdt * g0r[e];
    }

    // ======== stage 1: f1=drift(H01), g1=diff(H11), fused ========
    stage(sA, tmpA);
    stage(sAd, tmpB);
    __syncthreads();
    gemm_reg<128, 256, 0>(sA, BW1, bs1, sH, nullptr);
    gemm_reg<128, 256, 0>(sAd, BV1, cs1, sHd, nullptr);
    __syncthreads();
    gemm_glb<256, 256, 0>(sH, W2p, bs2, sH2, nullptr);
    gemm_reg<256, 128, 2>(sHd, BV2, cs2, nullptr, sG);
    __syncthreads();
    gemm_reg<256, 128, 1>(sH2, BW3, bs3, nullptr, sF);
    __syncthreads();
#pragma unroll
    for (int e = 0; e < EPT; ++e) {
      g1r[e] = sG[base + e];
      float gw = (4.f / 3.f) * IkR[e] - (4.f / 3.f) * IkkR[e] * inv_sqdt -
                 (4.f / 3.f) * Ik0R[e] * inv_dt + (5.f / 3.f) * IkkkR[e] * inv_dt;
      y1r[e] += (dt * (2.f / 3.f)) * sF[base + e] + gw * g1r[e];
      tmpA[e] = y0r[e] + dt * f0r[e] - sqdt * g0r[e];
    }

    // ======== stage 2: H02=y0 => f2=f0, ALPHA[2]=0; g2=diff(H12) ========
    stage(sA, tmpA);
    __syncthreads();
    gemm_reg<128, 256, 0>(sA, BV1, cs1, sHd, nullptr);
    __syncthreads();
    gemm_reg<256, 128, 2>(sHd, BV2, cs2, nullptr, sG);
    __syncthreads();
#pragma unroll
    for (int e = 0; e < EPT; ++e) {
      g2r[e] = sG[base + e];
      float gw = (2.f / 3.f) * IkR[e] + (1.f / 3.f) * IkkR[e] * inv_sqdt -
                 (2.f / 3.f) * Ik0R[e] * inv_dt - (2.f / 3.f) * IkkkR[e] * inv_dt;
      y1r[e] += gw * g2r[e];
      tmpA[e] = y0r[e] + 0.25f * dt * f0r[e] +
                sqdt * (-5.f * g0r[e] + 3.f * g1r[e] + 0.5f * g2r[e]);
    }

    // ======== stage 3: H03=y0 => f3=f0, ALPHA[3]=0; g3=diff(H13) ========
    stage(sA, tmpA);
    __syncthreads();
    gemm_reg<128, 256, 0>(sA, BV1, cs1, sHd, nullptr);
    __syncthreads();
    gemm_reg<256, 128, 2>(sHd, BV2, cs2, nullptr, sG);
    __syncthreads();
#pragma unroll
    for (int e = 0; e < EPT; ++e) {
      y1r[e] += IkkkR[e] * inv_dt * sG[base + e];
      y0r[e] = y1r[e];
    }

    if (((t + 1) % NSUB) == 0) {
      int k = (t + 1) / NSUB;
#pragma unroll
      for (int e = 0; e < EPT; ++e) {
        out[(size_t)(rowBase * 128 + base + e) * 10 + k] = y0r[e];
      }
    }
  }
}

extern "C" void kernel_launch(void* const* d_in, const int* in_sizes, int n_in,
                              void* d_out, int out_size, void* d_ws, size_t ws_size,
                              hipStream_t stream) {
  const float* x  = (const float*)d_in[0];
  const float* W1 = (const float*)d_in[1];
  const float* b1 = (const float*)d_in[2];
  const float* W2 = (const float*)d_in[3];
  const float* b2 = (const float*)d_in[4];
  const float* W3 = (const float*)d_in[5];
  const float* b3 = (const float*)d_in[6];
  const float* V1 = (const float*)d_in[7];
  const float* c1 = (const float*)d_in[8];
  const float* V2 = (const float*)d_in[9];
  const float* c2 = (const float*)d_in[10];
  const float* zw = (const float*)d_in[11];
  const float* zl = (const float*)d_in[12];

  unsigned* ws  = (unsigned*)d_ws;
  unsigned* W1p = ws;              // 128x256 -> 16384 dw
  unsigned* W2p = ws + 16384;      // 256x256 -> 32768 dw
  unsigned* W3p = ws + 49152;      // 256x128 -> 16384 dw
  unsigned* V1p = ws + 65536;      // 128x256 -> 16384 dw
  unsigned* V2p = ws + 81920;      // 256x128 -> 16384 dw

  pack_weight<<<64, 256, 0, stream>>>(W1, W1p, 128, 256);
  pack_weight<<<128, 256, 0, stream>>>(W2, W2p, 256, 256);
  pack_weight<<<64, 256, 0, stream>>>(W3, W3p, 256, 128);
  pack_weight<<<64, 256, 0, stream>>>(V1, V1p, 128, 256);
  pack_weight<<<64, 256, 0, stream>>>(V2, V2p, 256, 128);

  sde_kernel<<<64, 256, 0, stream>>>(x, b1, b2, b3, c1, c2, zw, zl,
                                     W1p, W2p, W3p, V1p, V2p, (float*)d_out);
}
